// SpikformerMLP_36756330119638
// MI455X (gfx1250) — matrix operand order repaired and checked by reading, not compiled, not run on hardware
//
#include <hip/hip_runtime.h>

typedef __attribute__((ext_vector_type(16))) __bf16 v16bf;
typedef __attribute__((ext_vector_type(8)))  float  v8f;

union Frag16 {
  unsigned int u[8];
  uint4 q[2];
  v16bf v;
};

__device__ __forceinline__ unsigned int bfpack(float a, float b) {
  unsigned int ua = __float_as_uint(a); ua += 0x7FFFu + ((ua >> 16) & 1u);
  unsigned int ub = __float_as_uint(b); ub += 0x7FFFu + ((ub >> 16) & 1u);
  return (ua >> 16) | (ub & 0xFFFF0000u);
}

__device__ __forceinline__ unsigned short bf16of(float a) {
  unsigned int ua = __float_as_uint(a); ua += 0x7FFFu + ((ua >> 16) & 1u);
  return (unsigned short)(ua >> 16);
}

#define T_   4
#define B_   32
#define C_   384
#define H_   1536
#define O_   384
#define N_   196
#define NP_  224    // padded N: 14 tiles of 16 -> 7 tile-pairs, no guards in GEMM
#define KC_  192    // K-chunk staged in LDS per buffer
#define KCP_ 200    // padded LDS row stride (elements): 400B = 100 dwords, gcd(100,64)=4
                    // -> 32 lanes' b128 reads tile all 64 banks conflict-free

// ---------------------------------------------------------------------------
// Kernel 0a: fp32 -> bf16 weight conversion
// ---------------------------------------------------------------------------
__global__ __launch_bounds__(256) void cvt_bf16_kernel(
    const float* __restrict__ src, unsigned short* __restrict__ dst, int nElem) {
  int i = blockIdx.x * blockDim.x + threadIdx.x;
  if (i < nElem) dst[i] = bf16of(src[i]);
}

// ---------------------------------------------------------------------------
// Kernel 0b: x[t,b,c,n] (f32) -> xT[t,b,n(pad 224),c] (bf16), zero-pad n>=196
// ---------------------------------------------------------------------------
__global__ __launch_bounds__(384) void xpose_kernel(
    const float* __restrict__ x, unsigned short* __restrict__ xT) {
  int n  = blockIdx.x;   // 0..223
  int tb = blockIdx.y;   // 0..127
  int c  = threadIdx.x;  // 0..383
  float val = (n < N_) ? x[(tb * C_ + c) * N_ + n] : 0.0f;
  xT[(tb * NP_ + n) * C_ + c] = bf16of(val);
}

// ---------------------------------------------------------------------------
// Async staging: one K-chunk of the B panel (4t x 32n x KC_ bf16) -> LDS.
// 256 threads: 2 threads/row, 12 x 16B async copies each (row = 384B).
// Uses the CDNA5 async-to-LDS path; completion tracked by ASYNCcnt.
// ---------------------------------------------------------------------------
template <int KROW>
__device__ __forceinline__ void stage_chunk(
    const unsigned short* __restrict__ src, int b, int n0, int kb,
    unsigned ldsBase, int tid) {
  const int row = tid >> 1;   // 0..127  (= t*32 + nn)
  const int seg = tid & 1;    // 0..1
  const int t  = row >> 5;
  const int nn = row & 31;
  const unsigned short* g =
      src + (size_t)((t * B_ + b) * NP_ + n0 + nn) * KROW + kb + seg * 8;
  unsigned l = ldsBase + (unsigned)(row * (KCP_ * 2) + seg * 16);
#pragma unroll
  for (int s = 0; s < 12; ++s) {
    asm volatile("global_load_async_to_lds_b128 %0, %1, off"
                 :: "v"(l + (unsigned)(s * 32)), "v"(g + s * 16)
                 : "memory");
  }
}

__device__ __forceinline__ void async_wait_sync() {
  asm volatile("s_wait_asynccnt 0x0" ::: "memory");
  __syncthreads();
}

// ---------------------------------------------------------------------------
// Kernel 1: GEMM1 (w1 x xT) + BN1 + LIF over T -> spikes s1T[t,b,n,h] bf16
// Block: 8 waves share one 32-column n-panel (staged in LDS, double-buffered);
// each wave owns one 16-row h-tile x two 16-col n-tiles for all 4 t.
// ---------------------------------------------------------------------------
__global__ __launch_bounds__(256) void mlp1_kernel(
    const unsigned short* __restrict__ w1b,
    const float* __restrict__ g1, const float* __restrict__ b1,
    const float* __restrict__ m1, const float* __restrict__ v1,
    const unsigned short* __restrict__ xT,
    unsigned short* __restrict__ s1t) {
  __shared__ unsigned short sB[2][128 * KCP_];

  const int tid  = threadIdx.x;
  const int lane = tid & 31;
  const int wave = tid >> 5;
  const int r    = lane & 15;
  const int half = lane >> 4;
  const int gpr  = blockIdx.x;               // n-tile pair 0..6
  const int htile = blockIdx.y * 8 + wave;   // 0..95
  const int b    = blockIdx.z;               // 0..31
  const int hbase = htile * 16;
  const int n0    = gpr * 32;

  unsigned lds0 = (unsigned)(unsigned long long)&sB[0][0];
  unsigned lds1 = (unsigned)(unsigned long long)&sB[1][0];

  float sc[8], sh[8];
#pragma unroll
  for (int j = 0; j < 8; ++j) {
    int h = hbase + half * 8 + j;
    float s = g1[h] * rsqrtf(v1[h] + 1e-5f);
    sc[j] = s;
    sh[j] = b1[h] - m1[h] * s;
  }

  v8f acc[2][4] = {};
  const unsigned short* arow = w1b + (hbase + r) * C_;

  stage_chunk<C_>(xT, b, n0, 0, lds0, tid);
  async_wait_sync();

  const int NCH = C_ / KC_;  // 2
  for (int ch = 0; ch < NCH; ++ch) {
    if (ch + 1 < NCH)
      stage_chunk<C_>(xT, b, n0, (ch + 1) * KC_, ((ch + 1) & 1) ? lds1 : lds0, tid);
    const unsigned short* sb = &sB[ch & 1][0];
    for (int kl = 0; kl < KC_; kl += 32) {
      const int kb = ch * KC_ + kl;
      Frag16 A;
      A.q[0] = *(const uint4*)(arow + kb + half * 8);
      A.q[1] = *(const uint4*)(arow + kb + half * 8 + 16);
#pragma unroll
      for (int t = 0; t < 4; ++t) {
#pragma unroll
        for (int nt = 0; nt < 2; ++nt) {
          Frag16 Bf;
          const unsigned short* bp =
              sb + (t * 32 + nt * 16 + r) * KCP_ + kl + half * 8;
          Bf.q[0] = *(const uint4*)(bp);
          Bf.q[1] = *(const uint4*)(bp + 16);
          acc[nt][t] = __builtin_amdgcn_wmma_f32_16x16x32_bf16(
              false, A.v, false, Bf.v, (short)0, acc[nt][t], false, false);
        }
      }
    }
    async_wait_sync();
  }

  // BN + LIF scan over t, then spikes -> s1T[t,b,n,h] (one b128 store per t,nt)
#pragma unroll
  for (int nt = 0; nt < 2; ++nt) {
    const int n = n0 + nt * 16 + r;
    float sp[4][8];
#pragma unroll
    for (int j = 0; j < 8; ++j) {
      float vm = 0.f;
#pragma unroll
      for (int t = 0; t < 4; ++t) {
        float y = acc[nt][t][j] * sc[j] + sh[j];
        vm = (vm + y) * 0.5f;
        float s = (vm >= 1.0f) ? 1.0f : 0.0f;
        sp[t][j] = s;
        vm = (1.0f - s) * vm;
      }
    }
#pragma unroll
    for (int t = 0; t < 4; ++t) {
      uint4 p;
      p.x = bfpack(sp[t][0], sp[t][1]);
      p.y = bfpack(sp[t][2], sp[t][3]);
      p.z = bfpack(sp[t][4], sp[t][5]);
      p.w = bfpack(sp[t][6], sp[t][7]);
      *(uint4*)(s1t + (size_t)((t * B_ + b) * NP_ + n) * H_ + hbase + half * 8) = p;
    }
  }
}

// ---------------------------------------------------------------------------
// Kernel 2: GEMM2 (w2 x s1T) + BN2 + LIF over T -> f32 spikes out[t,b,o,n]
// ---------------------------------------------------------------------------
__global__ __launch_bounds__(256) void mlp2_kernel(
    const unsigned short* __restrict__ w2b,
    const float* __restrict__ g2, const float* __restrict__ b2,
    const float* __restrict__ m2, const float* __restrict__ v2,
    const unsigned short* __restrict__ s1t,
    float* __restrict__ out) {
  __shared__ unsigned short sB[2][128 * KCP_];

  const int tid  = threadIdx.x;
  const int lane = tid & 31;
  const int wave = tid >> 5;
  const int r    = lane & 15;
  const int half = lane >> 4;
  const int gpr  = blockIdx.x;               // n-tile pair 0..6
  const int otile = blockIdx.y * 8 + wave;   // 0..23
  const int b    = blockIdx.z;               // 0..31
  const int obase = otile * 16;
  const int n0    = gpr * 32;

  unsigned lds0 = (unsigned)(unsigned long long)&sB[0][0];
  unsigned lds1 = (unsigned)(unsigned long long)&sB[1][0];

  float sc[8], sh[8];
#pragma unroll
  for (int j = 0; j < 8; ++j) {
    int o = obase + half * 8 + j;
    float s = g2[o] * rsqrtf(v2[o] + 1e-5f);
    sc[j] = s;
    sh[j] = b2[o] - m2[o] * s;
  }

  v8f acc[2][4] = {};
  const unsigned short* arow = w2b + (obase + r) * H_;

  stage_chunk<H_>(s1t, b, n0, 0, lds0, tid);
  async_wait_sync();

  const int NCH = H_ / KC_;  // 8
  for (int ch = 0; ch < NCH; ++ch) {
    if (ch + 1 < NCH)
      stage_chunk<H_>(s1t, b, n0, (ch + 1) * KC_, ((ch + 1) & 1) ? lds1 : lds0, tid);
    const unsigned short* sb = &sB[ch & 1][0];
    for (int kl = 0; kl < KC_; kl += 32) {
      const int kb = ch * KC_ + kl;
      Frag16 A;
      A.q[0] = *(const uint4*)(arow + kb + half * 8);
      A.q[1] = *(const uint4*)(arow + kb + half * 8 + 16);
#pragma unroll
      for (int t = 0; t < 4; ++t) {
#pragma unroll
        for (int nt = 0; nt < 2; ++nt) {
          Frag16 Bf;
          const unsigned short* bp =
              sb + (t * 32 + nt * 16 + r) * KCP_ + kl + half * 8;
          Bf.q[0] = *(const uint4*)(bp);
          Bf.q[1] = *(const uint4*)(bp + 16);
          acc[nt][t] = __builtin_amdgcn_wmma_f32_16x16x32_bf16(
              false, A.v, false, Bf.v, (short)0, acc[nt][t], false, false);
        }
      }
    }
    async_wait_sync();
  }

#pragma unroll
  for (int nt = 0; nt < 2; ++nt) {
    const int n = n0 + nt * 16 + r;
    float sp[4][8];
#pragma unroll
    for (int j = 0; j < 8; ++j) {
      float vm = 0.f;
#pragma unroll
      for (int t = 0; t < 4; ++t) {
        float y = acc[nt][t][j] * sc[j] + sh[j];
        vm = (vm + y) * 0.5f;
        float s = (vm >= 1.0f) ? 1.0f : 0.0f;
        sp[t][j] = s;
        vm = (1.0f - s) * vm;
      }
    }
    if (n < N_) {
#pragma unroll
      for (int t = 0; t < 4; ++t) {
#pragma unroll
        for (int j = 0; j < 8; ++j) {
          int o = obase + half * 8 + j;
          out[(size_t)((t * B_ + b) * O_ + o) * N_ + n] = sp[t][j];
        }
      }
    }
  }
}

// ---------------------------------------------------------------------------
// Launch
// ---------------------------------------------------------------------------
extern "C" void kernel_launch(void* const* d_in, const int* in_sizes, int n_in,
                              void* d_out, int out_size, void* d_ws, size_t ws_size,
                              hipStream_t stream) {
  const float* x  = (const float*)d_in[0];
  const float* w1 = (const float*)d_in[1];
  const float* g1 = (const float*)d_in[2];
  const float* b1 = (const float*)d_in[3];
  const float* m1 = (const float*)d_in[4];
  const float* v1 = (const float*)d_in[5];
  const float* w2 = (const float*)d_in[6];
  const float* g2 = (const float*)d_in[7];
  const float* b2 = (const float*)d_in[8];
  const float* m2 = (const float*)d_in[9];
  const float* v2 = (const float*)d_in[10];

  // Workspace layout (bytes):
  //   xT  : T*B*NP*C bf16 = 22,020,096
  //   s1T : T*B*NP*H bf16 = 88,080,384
  //   w1b : H*C      bf16 =  1,179,648
  //   w2b : O*H      bf16 =  1,179,648
  char* ws = (char*)d_ws;
  unsigned short* xT  = (unsigned short*)(ws);
  unsigned short* s1T = (unsigned short*)(ws + 22020096);
  unsigned short* w1b = (unsigned short*)(ws + 22020096 + 88080384);
  unsigned short* w2b = (unsigned short*)(ws + 22020096 + 88080384 + 1179648);

  const int nw = H_ * C_;  // 589824 for both weight matrices
  cvt_bf16_kernel<<<(nw + 255) / 256, 256, 0, stream>>>(w1, w1b, nw);
  cvt_bf16_kernel<<<(nw + 255) / 256, 256, 0, stream>>>(w2, w2b, nw);

  xpose_kernel<<<dim3(NP_, T_ * B_), 384, 0, stream>>>(x, xT);

  // 7 n-tile-pairs x (96 h-tiles / 8 waves) x 32 batch
  mlp1_kernel<<<dim3(7, 12, 32), 256, 0, stream>>>(w1b, g1, b1, m1, v1, xT, s1T);

  // 7 n-tile-pairs x (24 o-tiles / 8 waves) x 32 batch
  mlp2_kernel<<<dim3(7, 3, 32), 256, 0, stream>>>(w2b, g2, b2, m2, v2, s1T, (float*)d_out);
}
